// PTV3_CPE_38371237822879
// MI455X (gfx1250) — matrix operand order repaired and checked by reading, not compiled, not run on hardware
//
#include <hip/hip_runtime.h>
#include <cstddef>
#include <cstdint>

typedef __attribute__((ext_vector_type(16))) __bf16 v16bf;
typedef __attribute__((ext_vector_type(4)))  __bf16 bf16x4;
typedef __attribute__((ext_vector_type(8)))  float  v8f;
typedef __attribute__((ext_vector_type(4)))  float  f32x4;
typedef __attribute__((ext_vector_type(4)))  unsigned int u32x4;

static constexpr int CH   = 128;  // channels
static constexpr int KV   = 27;   // kernel volume
static constexpr int LSTR = 136;  // LDS row stride in bf16 elems (272B: 16B-aligned, bank-rotating)

union FragU {
    v16bf v;
    u32x4 q[2];
};

__device__ __forceinline__ __bf16 f2b(float f) { return (__bf16)f; }

// ---- gfx1250 async global->LDS copy (16B per lane), tracked by ASYNCcnt ----
__device__ __forceinline__ void async_copy16(const __bf16* gsrc, __bf16* ldst)
{
    // Generic pointer to __shared__ has the wave-relative LDS byte address in its low 32 bits.
    unsigned lds_addr = (unsigned)(uintptr_t)ldst;
    unsigned long long gaddr = (unsigned long long)(uintptr_t)gsrc;
    asm volatile("global_load_async_to_lds_b128 %0, %1, off"
                 :: "v"(lds_addr), "v"(gaddr) : "memory");
}
__device__ __forceinline__ void wait_async()
{
    asm volatile("s_wait_asynccnt 0" ::: "memory");
}

// ================= pre-pass: f32 -> bf16 conversions into workspace =================
__global__ void cvt_feats_kernel(const float* __restrict__ f, __bf16* __restrict__ o, long long n4)
{
    long long i = (long long)blockIdx.x * blockDim.x + threadIdx.x;
    if (i < n4) {
        f32x4 v = ((const f32x4*)f)[i];
        bf16x4 r = { f2b(v[0]), f2b(v[1]), f2b(v[2]), f2b(v[3]) };
        ((bf16x4*)o)[i] = r;
    }
}

__global__ void cvt_weights_kernel(const float* __restrict__ conv_w, const float* __restrict__ lin_w,
                                   __bf16* __restrict__ wt, __bf16* __restrict__ lwt)
{
    int i = blockIdx.x * blockDim.x + threadIdx.x;   // 27*16384 + 16384 threads
    if (i < KV * CH * CH) {
        // transpose conv_w[k][cin][cout] -> wt[k][cout][cin]
        const int k = i >> 14, r = i & 16383, cout = r >> 7, cin = r & 127;
        wt[i] = f2b(conv_w[(size_t)k * CH * CH + cin * CH + cout]);
    } else {
        const int j = i - KV * CH * CH;
        lwt[j] = f2b(lin_w[j]);                       // lin_w[cout][cin] is already B^T layout
    }
}

// ================= fused conv + linear + layernorm =================
// PRE=true : bf16 feats/weights in workspace, async double-buffered weight staging.
// PRE=false: self-contained f32 path (round-1 behavior), single LDS buffer.
template <bool PRE>
__global__ __launch_bounds__(256, 2)
void ptv3_fused_kernel(const float*  __restrict__ feats_f32,
                       const __bf16* __restrict__ feats_bf,
                       const int*    __restrict__ nbr,
                       const float*  __restrict__ conv_w,
                       const __bf16* __restrict__ wt,
                       const float*  __restrict__ conv_b,
                       const float*  __restrict__ lin_w,
                       const __bf16* __restrict__ lwt,
                       const float*  __restrict__ lin_b,
                       const float*  __restrict__ ln_g,
                       const float*  __restrict__ ln_b,
                       float* __restrict__ out,
                       int nrows)
{
    extern __shared__ __align__(16) char smem_raw[];
    __bf16* wbuf0 = (__bf16*)smem_raw;             // 34,816 B
    __bf16* wbuf1 = wbuf0 + CH * LSTR;             // second buffer (PRE only)

    const int tid  = threadIdx.x;
    const int wave = tid >> 5;          // 8 waves; wave w owns rows [16w, 16w+16)
    const int l16  = tid & 15;
    const int sel  = (tid >> 4) & 1;    // 0 = lanes 0-15, 1 = lanes 16-31

    const int block0 = blockIdx.x * 128;
    const int myrow  = block0 + wave * 16 + l16;
    const int rowcl  = myrow < nrows ? myrow : nrows - 1;

    const v8f vzero = {};
    v8f acc[8];
#pragma unroll
    for (int n = 0; n < 8; ++n) acc[n] = vzero;

    // async stage of pre-converted bf16 conv_w[k]^T tile into an LDS buffer (with row padding)
    auto stage_async = [&](int kidx, __bf16* dst) {
        const __bf16* src = wt + (size_t)kidx * CH * CH;
#pragma unroll
        for (int i = 0; i < 8; ++i) {
            const int c    = tid + 256 * i;        // 16B chunk id, 2048 chunks
            const int cout = c >> 4;
            const int cin0 = (c & 15) << 3;
            async_copy16(src + c * 8, dst + cout * LSTR + cin0);
        }
    };

    if constexpr (PRE) {
        stage_async(0, wbuf0);
        wait_async();
    }
    __syncthreads();

    // ================= 27-tap gather conv: acc += gather(feats,k) @ conv_w[k] =================
#pragma unroll 1
    for (int k = 0; k < KV; ++k) {
        const __bf16* cur;
        if constexpr (PRE) {
            cur = (k & 1) ? wbuf1 : wbuf0;
            if (k + 1 < KV)                         // DMA next tap's weights during this GEMM
                stage_async(k + 1, (k & 1) ? wbuf0 : wbuf1);
        } else {
            cur = wbuf0;
            // stage conv_w[k] (f32 [cin][cout]) transposed into LDS bf16 [cout][cin]
            const float* wk = conv_w + (size_t)k * CH * CH;
#pragma unroll
            for (int i = 0; i < 16; ++i) {
                const int lin  = 4 * tid + 1024 * i;
                const int cin  = lin >> 7;
                const int cout = lin & 127;
                const f32x4 wv = *(const f32x4*)(wk + lin);
                wbuf0[(cout + 0) * LSTR + cin] = f2b(wv[0]);
                wbuf0[(cout + 1) * LSTR + cin] = f2b(wv[1]);
                wbuf0[(cout + 2) * LSTR + cin] = f2b(wv[2]);
                wbuf0[(cout + 3) * LSTR + cin] = f2b(wv[3]);
            }
            __syncthreads();
            if (k + 1 < KV)
                __builtin_prefetch(conv_w + (size_t)(k + 1) * CH * CH + tid * 64, 0, 3);
        }

        // per-lane neighbor row (A-matrix gather); invalid neighbor -> zeros
        const int  idx   = nbr[(size_t)rowcl * KV + k];
        const bool valid = idx >= 0;
        const size_t rsafe = (size_t)(valid ? idx : 0) * CH;

#pragma unroll
        for (int kk = 0; kk < 4; ++kk) {
            // 16-bit A 16x32 layout: lane<16 holds K{o..o+7, o+16..o+23}, o=kk*32 (+8 for sel=1)
            const int o0 = kk * 32 + sel * 8;
            v16bf a;
            if constexpr (PRE) {
                const __bf16* rowp = feats_bf + rsafe;
                const u32x4 zq = {};
                u32x4 q0 = *(const u32x4*)(rowp + o0);
                u32x4 q1 = *(const u32x4*)(rowp + o0 + 16);
                FragU af;
                af.q[0] = valid ? q0 : zq;
                af.q[1] = valid ? q1 : zq;
                a = af.v;
            } else {
                const float* rowp = feats_f32 + rsafe;
                const float scale = valid ? 1.0f : 0.0f;
                const f32x4 a0 = *(const f32x4*)(rowp + o0);
                const f32x4 a1 = *(const f32x4*)(rowp + o0 + 4);
                const f32x4 a2 = *(const f32x4*)(rowp + o0 + 16);
                const f32x4 a3 = *(const f32x4*)(rowp + o0 + 20);
#pragma unroll
                for (int i = 0; i < 4; ++i) {
                    a[i]      = f2b(a0[i] * scale);
                    a[4 + i]  = f2b(a1[i] * scale);
                    a[8 + i]  = f2b(a2[i] * scale);
                    a[12 + i] = f2b(a3[i] * scale);
                }
            }
#pragma unroll
            for (int n = 0; n < 8; ++n) {
                // B 32x16: lane = column N, K{o..o+7, o+16..o+23} (o=kk*32, +8 for sel=1) from transposed LDS row
                const __bf16* bp = cur + (n * 16 + l16) * LSTR + kk * 32 + sel * 8;
                FragU b;
                b.q[0] = *(const u32x4*)(bp);
                b.q[1] = *(const u32x4*)(bp + 16);
                acc[n] = __builtin_amdgcn_wmma_f32_16x16x32_bf16(
                             false, a, false, b.v, (short)0, acc[n], false, false);
            }
        }

        if constexpr (PRE) { wait_async(); }
        __syncthreads();   // weight buffer handoff for next iteration
    }

    // ================= + conv bias =================
#pragma unroll
    for (int t = 0; t < 8; ++t) {
        const float cb = conv_b[l16 + 16 * t];
#pragma unroll
        for (int j = 0; j < 8; ++j) acc[t][j] += cb;
    }

    // ================= restage conv result (bf16) into LDS as next A-tile =================
    {
        const int rbase = wave * 16 + 8 * sel;  // C/D layout: VGPR j -> row j (+8 for upper lanes)
#pragma unroll
        for (int t = 0; t < 8; ++t)
#pragma unroll
            for (int j = 0; j < 8; ++j)
                wbuf0[(rbase + j) * LSTR + l16 + 16 * t] = f2b(acc[t][j]);
    }
    __syncthreads();

    // ================= GEMM2: acc = conv_result @ lin_w^T =================
#pragma unroll
    for (int n = 0; n < 8; ++n) acc[n] = vzero;

#pragma unroll
    for (int kk = 0; kk < 4; ++kk) {
        const __bf16* ap = wbuf0 + (wave * 16 + l16) * LSTR + kk * 32 + sel * 8;
        FragU a;
        a.q[0] = *(const u32x4*)(ap);
        a.q[1] = *(const u32x4*)(ap + 16);
#pragma unroll
        for (int n = 0; n < 8; ++n) {
            v16bf b;
            if constexpr (PRE) {
                const __bf16* wp = lwt + (size_t)(n * 16 + l16) * CH + kk * 32 + sel * 8;
                FragU bf;
                bf.q[0] = *(const u32x4*)(wp);
                bf.q[1] = *(const u32x4*)(wp + 16);
                b = bf.v;
            } else {
                const float* wp = lin_w + (size_t)(n * 16 + l16) * CH + kk * 32 + sel * 8;
                const f32x4 b0 = *(const f32x4*)(wp);
                const f32x4 b1 = *(const f32x4*)(wp + 4);
                const f32x4 b2 = *(const f32x4*)(wp + 16);
                const f32x4 b3 = *(const f32x4*)(wp + 20);
#pragma unroll
                for (int i = 0; i < 4; ++i) {
                    b[i]      = f2b(b0[i]);
                    b[4 + i]  = f2b(b1[i]);
                    b[8 + i]  = f2b(b2[i]);
                    b[12 + i] = f2b(b3[i]);
                }
            }
            acc[n] = __builtin_amdgcn_wmma_f32_16x16x32_bf16(
                         false, a.v, false, b, (short)0, acc[n], false, false);
        }
    }

    // ================= + lin bias, LayerNorm (shuffle reduction), store =================
    float gv[8], bvv[8], lb[8];
#pragma unroll
    for (int t = 0; t < 8; ++t) {
        const int col = l16 + 16 * t;
        lb[t]  = lin_b[col];
        gv[t]  = ln_g[col];
        bvv[t] = ln_b[col];
    }

#pragma unroll
    for (int j = 0; j < 8; ++j) {
        float s = 0.f, q = 0.f;
        float v[8];
#pragma unroll
        for (int t = 0; t < 8; ++t) {
            v[t] = acc[t][j] + lb[t];
            s += v[t];
            q += v[t] * v[t];
        }
        // row spans 16 lanes (one wave half); xor masks < 16 stay within the half
        s += __shfl_xor(s, 8, 32); q += __shfl_xor(q, 8, 32);
        s += __shfl_xor(s, 4, 32); q += __shfl_xor(q, 4, 32);
        s += __shfl_xor(s, 2, 32); q += __shfl_xor(q, 2, 32);
        s += __shfl_xor(s, 1, 32); q += __shfl_xor(q, 1, 32);
        const float mu   = s * (1.0f / 128.0f);
        const float var  = q * (1.0f / 128.0f) - mu * mu;
        const float rstd = rsqrtf(var + 1e-5f);

        const int grow = block0 + wave * 16 + j + 8 * sel;
        if (grow < nrows) {
            float* orow = out + (size_t)grow * CH;
#pragma unroll
            for (int t = 0; t < 8; ++t)
                orow[l16 + 16 * t] = (v[t] - mu) * rstd * gv[t] + bvv[t];
        }
    }
}

extern "C" void kernel_launch(void* const* d_in, const int* in_sizes, int n_in,
                              void* d_out, int out_size, void* d_ws, size_t ws_size,
                              hipStream_t stream)
{
    const float* feats  = (const float*)d_in[0];
    const int*   nbr    = (const int*)  d_in[1];
    const float* conv_w = (const float*)d_in[2];
    const float* conv_b = (const float*)d_in[3];
    const float* lin_w  = (const float*)d_in[4];
    const float* lin_b  = (const float*)d_in[5];
    const float* ln_g   = (const float*)d_in[6];
    const float* ln_b   = (const float*)d_in[7];
    (void)n_in; (void)out_size;

    const int nrows  = in_sizes[0] / CH;      // 400000
    const int blocks = (nrows + 127) / 128;   // 3125

    const size_t nfeat = (size_t)in_sizes[0];                 // N*C elements
    const size_t need  = nfeat * 2 + (size_t)KV * CH * CH * 2 + (size_t)CH * CH * 2;
    const bool   pre   = (ws_size >= need) && (((uintptr_t)d_ws & 15) == 0);

    if (pre) {
        __bf16* feats_bf = (__bf16*)d_ws;
        __bf16* wt       = feats_bf + nfeat;
        __bf16* lwt      = wt + (size_t)KV * CH * CH;

        const long long n4 = (long long)(nfeat / 4);
        cvt_feats_kernel<<<(int)((n4 + 255) / 256), 256, 0, stream>>>(feats, feats_bf, n4);
        const int wtot = KV * CH * CH + CH * CH;
        cvt_weights_kernel<<<(wtot + 255) / 256, 256, 0, stream>>>(conv_w, lin_w, wt, lwt);

        ptv3_fused_kernel<true><<<blocks, 256, 2 * CH * LSTR * sizeof(__bf16), stream>>>(
            feats, feats_bf, nbr, conv_w, wt, conv_b, lin_w, lwt, lin_b, ln_g, ln_b,
            (float*)d_out, nrows);
    } else {
        ptv3_fused_kernel<false><<<blocks, 256, CH * LSTR * sizeof(__bf16), stream>>>(
            feats, nullptr, nbr, conv_w, nullptr, conv_b, lin_w, nullptr, lin_b, ln_g, ln_b,
            (float*)d_out, nrows);
    }
}
